// MultiHeadAttention_63436666962234
// MI455X (gfx1250) — matrix operand order repaired and checked by reading, not compiled, not run on hardware
//
#include <hip/hip_runtime.h>

// ---------------------------------------------------------------------------
// MHA with per-head full-width projections on gfx1250 (MI455X).
// All matmuls via v_wmma_f32_16x16x32_f16 (f16 in, f32 accumulate).
// Flash-attention streaming keeps the 134MB score matrix out of HBM,
// using the 320KB/WGP LDS for a transposed 128x512 V tile.
// Data movement uses the CDNA5-specific paths:
//   - TDM tensor_load_to_lds (TENSORcnt) for the padded 2D Q-tile load
//   - global_load_async_to_lds_b128 (ASYNCcnt) for f16 GEMM tile staging
//   - global_prefetch for next-tile warmup
// ---------------------------------------------------------------------------

typedef _Float16 h16;
typedef __attribute__((ext_vector_type(8)))  _Float16 v8h;
typedef __attribute__((ext_vector_type(16))) _Float16 v16h;
typedef __attribute__((ext_vector_type(8)))  float    v8f;
typedef __attribute__((ext_vector_type(4)))  unsigned int u32x4;
typedef __attribute__((ext_vector_type(8)))  unsigned int u32x8;

#define B_  2
#define S_  2048
#define D_  512
#define H_  8
#define M_  (B_*S_)      // 4096 rows (b*S+s)
#define HD_ (H_*D_)      // 4096 concat width

static __device__ __forceinline__ v8f wmma16(v16h a, v16h b, v8f c) {
  // D = A(16x32 f16) x B(32x16 f16) + C(16x16 f32)
  return __builtin_amdgcn_wmma_f32_16x16x32_f16(false, a, false, b, (short)0, c,
                                                false, false);
}
static __device__ __forceinline__ v16h cat16(v8h lo, v8h hi) {
  return __builtin_shufflevector(lo, hi, 0,1,2,3,4,5,6,7,8,9,10,11,12,13,14,15);
}

// CDNA5 async copy: 16B global -> LDS per lane, tracked by ASYNCcnt.
static __device__ __forceinline__ void async_ld_b128(void* lds, const void* gsrc) {
  asm volatile("global_load_async_to_lds_b128 %0, %1, off"
               :: "v"((unsigned)(unsigned long long)lds), "v"(gsrc)
               : "memory");
}
static __device__ __forceinline__ void wait_asynccnt0() {
  asm volatile("s_wait_asynccnt 0x0" ::: "memory");
}

// CDNA5 Tensor Data Mover: 2D tile load, D# groups 0 (4 SGPRs) + 1 (8 SGPRs).
static __device__ __forceinline__ void tdm_load_2d(u32x4 g0, u32x8 g1) {
  asm volatile("tensor_load_to_lds %0, %1" :: "s"(g0), "s"(g1) : "memory");
}
static __device__ __forceinline__ void wait_tensorcnt0() {
#if defined(__has_builtin) && __has_builtin(__builtin_amdgcn_s_wait_tensorcnt)
  __builtin_amdgcn_s_wait_tensorcnt(0);
#else
  asm volatile("s_wait_tensorcnt 0x0" ::: "memory");
#endif
}

// ---------------------------------------------------------------------------
// Kernel 1: batched projection GEMMs.  z = type*8 + head, type 0/1/2 = Q/K/V.
// Y[h][m][e] = (X[m][:] @ W[h][:][e] + b[h][e]) * (type==Q ? 1/sqrt(D) : 1)
// 128x128 tile per block, 8 waves, each wave owns 16 rows x 128 cols.
// (f32->f16 conversion happens in staging, so these stages stay synchronous.)
// ---------------------------------------------------------------------------
__global__ __launch_bounds__(256) void proj_gemm(
    const float* __restrict__ q, const float* __restrict__ k, const float* __restrict__ v,
    const float* __restrict__ Wq, const float* __restrict__ Wk, const float* __restrict__ Wv,
    const float* __restrict__ bq, const float* __restrict__ bk, const float* __restrict__ bv,
    h16* __restrict__ Qh, h16* __restrict__ Kh, h16* __restrict__ Vh)
{
  constexpr int BM = 128, BN = 128, BK = 32, PAD = 8;
  __shared__ alignas(16) h16 As[BM][BK + PAD];   // row-major A tile (f16)
  __shared__ alignas(16) h16 Bs[BN][BK + PAD];   // B tile stored transposed [n][k]

  const int tid = threadIdx.x, wave = tid >> 5, lane = tid & 31;
  const int bm = blockIdx.x * BM, bn = blockIdx.y * BN;
  const int z = blockIdx.z, type = z >> 3, h = z & 7;

  const float* X  = (type == 0) ? q : (type == 1) ? k : v;
  const float* W  = ((type == 0) ? Wq : (type == 1) ? Wk : Wv) + (size_t)h * D_ * D_;
  const float* bb = ((type == 0) ? bq : (type == 1) ? bk : bv) + (size_t)h * D_;
  h16* O = ((type == 0) ? Qh : (type == 1) ? Kh : Vh) + (size_t)h * M_ * D_;
  const float oscale = (type == 0) ? 0.04419417382415922f : 1.0f;  // 1/sqrt(512)

  v8f zero = {};
  v8f C[8];
#pragma unroll
  for (int g = 0; g < 8; ++g) C[g] = zero;

  for (int kk = 0; kk < D_; kk += BK) {
    // prefetch next k-step tiles while this one is staged/consumed
    if (kk + BK < D_) {
      __builtin_prefetch(&X[(size_t)(bm + (tid & 127)) * D_ + kk + BK + (tid >> 7) * 16], 0, 1);
      if (tid < BK) __builtin_prefetch(&W[(size_t)(kk + BK + tid) * D_ + bn], 0, 1);
    }
    // stage A (f32 -> f16)
    for (int idx = tid; idx < BM * (BK / 4); idx += 256) {
      const int r = idx >> 3, c = (idx & 7) * 4;
      float4 f = *reinterpret_cast<const float4*>(&X[(size_t)(bm + r) * D_ + kk + c]);
      As[r][c + 0] = (h16)f.x; As[r][c + 1] = (h16)f.y;
      As[r][c + 2] = (h16)f.z; As[r][c + 3] = (h16)f.w;
    }
    // stage B transposed (f32 -> f16), so B-fragments are contiguous in K
    for (int idx = tid; idx < BK * (BN / 4); idx += 256) {
      const int kr = idx >> 5, n = (idx & 31) * 4;
      float4 f = *reinterpret_cast<const float4*>(&W[(size_t)(kk + kr) * D_ + bn + n]);
      Bs[n + 0][kr] = (h16)f.x; Bs[n + 1][kr] = (h16)f.y;
      Bs[n + 2][kr] = (h16)f.z; Bs[n + 3][kr] = (h16)f.w;
    }
    __syncthreads();

    // A fragment: lane = 16*kb + m ; 16 contiguous K halves at kb*16
    const int mr = wave * 16 + (lane & 15), kb = lane >> 4;
    v8h alo = *reinterpret_cast<const v8h*>(&As[mr][kb * 16]);
    v8h ahi = *reinterpret_cast<const v8h*>(&As[mr][kb * 16 + 8]);
    v16h a = cat16(alo, ahi);
#pragma unroll
    for (int g = 0; g < 8; ++g) {
      // B fragment: lane = 16*kg + n ; 16 contiguous K halves at kg*16
      const int nr = g * 16 + (lane & 15), kg = (lane >> 4) * 16;
      v8h blo = *reinterpret_cast<const v8h*>(&Bs[nr][kg]);
      v8h bhi = *reinterpret_cast<const v8h*>(&Bs[nr][kg + 8]);
      C[g] = wmma16(a, cat16(blo, bhi), C[g]);
    }
    __syncthreads();
  }

  // epilogue: bias in f32, then scale, store f16
  const int ml = 8 * (lane >> 4);
#pragma unroll
  for (int g = 0; g < 8; ++g) {
    const int col = bn + g * 16 + (lane & 15);
    const float bias = bb[col];
#pragma unroll
    for (int vv = 0; vv < 8; ++vv) {
      const int row = bm + wave * 16 + ml + vv;   // C layout: m = vv + 8*(lane>=16)
      O[(size_t)row * D_ + col] = (h16)((C[g][vv] + bias) * oscale);
    }
  }
}

// ---------------------------------------------------------------------------
// Kernel 2: flash attention.  Block = 16 query rows of one (h,b).
// Q tile loaded by the Tensor Data Mover with native per-row LDS padding;
// scores via WMMA (K straight from global: contraction dim is contiguous),
// online softmax in LDS, then P @ V via WMMA with V staged transposed in LDS.
// ---------------------------------------------------------------------------
__global__ __launch_bounds__(256) void attn_flash(
    const h16* __restrict__ Qh, const h16* __restrict__ Kh, const h16* __restrict__ Vh,
    h16* __restrict__ concat)
{
  constexpr int KB = 128;
  constexpr int QS_LD = 520;   // 512 + 8 halves  (pad -> 4-bank rotation)
  constexpr int VST_LD = 136;  // 128 + 8 halves
  constexpr int SS_LD = 132;   // 128 + 4 floats
  constexpr int PS_LD = 136;   // 128 + 8 halves

  extern __shared__ char smem[];
  h16*   Qs    = (h16*)(smem);                     // [16][520]   16640 B
  h16*   Vst   = (h16*)(smem + 16640);             // [512][136] 139264 B (V^T)
  float* Ss    = (float*)(smem + 155904);          // [16][132]    8448 B
  h16*   Ps    = (h16*)(smem + 164352);            // [16][136]    4352 B
  float* red   = (float*)(smem + 168704);          // [16][16]     1024 B
  float* m_run = (float*)(smem + 169728);          // [16]
  float* l_run = m_run + 16;                       // [16]
  float* csc   = l_run + 16;                       // [16]

  const int tid = threadIdx.x, wave = tid >> 5, lane = tid & 31;
  const int qbase = blockIdx.x * 16;
  const int hb = blockIdx.y;                  // h*B + b
  const int h = hb >> 1, b = hb & 1;
  const size_t rowbase = (size_t)hb * S_;
  const h16* Qp = Qh + (rowbase + qbase) * D_;
  const h16* Kp = Kh + rowbase * D_;
  const h16* Vp = Vh + rowbase * D_;

  if (tid < 16) { m_run[tid] = -1e30f; l_run[tid] = 0.0f; }

  // ---- Q tile via TDM: 2D tile, 16 rows x 512 halves, +8 halves pad/row ----
  if (tid < 32) {
    const unsigned long long ga = (unsigned long long)(const void*)Qp;
    u32x4 g0;
    g0[0] = 1u;                                        // count=1 (valid user D#)
    g0[1] = (unsigned)(unsigned long long)(void*)Qs;   // lds_addr (bytes)
    g0[2] = (unsigned)ga;                              // global_addr[31:0]
    g0[3] = (unsigned)((ga >> 32) & 0x01FFFFFFu) | (2u << 30);  // [56:32] | type=2
    u32x8 g1;
    g1[0] = (1u << 16)        // data_size = 2 bytes
          | (1u << 20)        // pad_enable
          | (7u << 22)        // pad_interval = 256 DWORDs (one 512-half row)
          | (3u << 25);       // pad_amount   = 4 DWORDs   (8 halves)
    g1[1] = (512u & 0xFFFFu) << 16;   // tensor_dim0 = 512 (bits 79:48)
    g1[2] = (16u << 16);              // tensor_dim1 = 16  (bits 111:80)
    g1[3] = (512u << 16);             // tile_dim0  = 512  (bits 127:112)
    g1[4] = 16u;                      // tile_dim1 = 16, tile_dim2 = 0
    g1[5] = 512u;                     // tensor_dim0_stride = 512 (low 32)
    g1[6] = 0u;
    g1[7] = 0u;                       // tensor_dim1_stride unused (2D)
    tdm_load_2d(g0, g1);
  }
  wait_tensorcnt0();
  __syncthreads();

  v8f zero = {};
  v8f C[4];
#pragma unroll
  for (int g = 0; g < 4; ++g) C[g] = zero;

  for (int kb0 = 0; kb0 < S_; kb0 += KB) {
    // prefetch next key block (K and V) while working on this one
    if (kb0 + KB < S_) {
      const int pr = tid & 127, ph = tid >> 7;
      __builtin_prefetch(&Kp[(size_t)(kb0 + KB + pr) * D_ + ph * 256], 0, 1);
      __builtin_prefetch(&Vp[(size_t)(kb0 + KB + pr) * D_ + ph * 256], 0, 1);
    }
    // ---- stage V block transposed: Vst[e][key] (layout change -> manual) ----
    for (int idx = tid; idx < KB * 64; idx += 256) {
      const int kr = idx >> 6, c = (idx & 63) * 8;
      v8h vv = *reinterpret_cast<const v8h*>(&Vp[(size_t)(kb0 + kr) * D_ + c]);
#pragma unroll
      for (int j = 0; j < 8; ++j) Vst[(c + j) * VST_LD + kr] = vv[j];
    }

    // ---- phase 1: scores S = (Q*scale) . K^T ; wave -> cols [wave*16, +16) ----
    {
      v8f sc = zero;
      const int keyr = kb0 + wave * 16 + (lane & 15);
      const int kb = lane >> 4;
#pragma unroll
      for (int ks = 0; ks < 16; ++ks) {                 // 512 / 32
        v8h alo = *reinterpret_cast<const v8h*>(&Qs[(lane & 15) * QS_LD + ks * 32 + kb * 16]);
        v8h ahi = *reinterpret_cast<const v8h*>(&Qs[(lane & 15) * QS_LD + ks * 32 + kb * 16 + 8]);
        v8h blo = *reinterpret_cast<const v8h*>(&Kp[(size_t)keyr * D_ + ks * 32 + kb * 16]);
        v8h bhi = *reinterpret_cast<const v8h*>(&Kp[(size_t)keyr * D_ + ks * 32 + kb * 16 + 8]);
        sc = wmma16(cat16(alo, ahi), cat16(blo, bhi), sc);
      }
#pragma unroll
      for (int vv = 0; vv < 8; ++vv)
        Ss[(vv + 8 * (lane >> 4)) * SS_LD + wave * 16 + (lane & 15)] = sc[vv];
    }
    __syncthreads();

    // ---- phase 2: online softmax over this 16x128 strip ----
    {
      const int r = tid >> 4, seg = tid & 15;
      float pm = -1e30f;
#pragma unroll
      for (int j = 0; j < 8; ++j) pm = fmaxf(pm, Ss[r * SS_LD + seg * 8 + j]);
      red[r * 16 + seg] = pm;
      __syncthreads();
      if (tid < 16) {
        float bmax = -1e30f;
        for (int j = 0; j < 16; ++j) bmax = fmaxf(bmax, red[tid * 16 + j]);
        const float mo = m_run[tid], mn = fmaxf(mo, bmax);
        csc[tid] = __expf(mo - mn);          // first block: exp(-1e30) = 0
        m_run[tid] = mn;
      }
      __syncthreads();
      const float mn = m_run[r];
      float psum = 0.f;
#pragma unroll
      for (int j = 0; j < 8; ++j) {
        const float p = __expf(Ss[r * SS_LD + seg * 8 + j] - mn);
        Ps[r * PS_LD + seg * 8 + j] = (h16)p;
        psum += p;
      }
      red[r * 16 + seg] = psum;
      __syncthreads();
      if (tid < 16) {
        float s = 0.f;
        for (int j = 0; j < 16; ++j) s += red[tid * 16 + j];
        l_run[tid] = l_run[tid] * csc[tid] + s;
      }
      __syncthreads();
    }

    // ---- phase 3: O += P(16x128) @ V(128x64-slice) ; wave -> e-cols [wave*64, +64) ----
    {
      float cs[8];
#pragma unroll
      for (int vv = 0; vv < 8; ++vv) cs[vv] = csc[vv + 8 * (lane >> 4)];
#pragma unroll
      for (int g = 0; g < 4; ++g)
#pragma unroll
        for (int vv = 0; vv < 8; ++vv) C[g][vv] *= cs[vv];

      const int kb = lane >> 4;
#pragma unroll
      for (int kk = 0; kk < 4; ++kk) {                  // 128 / 32
        v8h alo = *reinterpret_cast<const v8h*>(&Ps[(lane & 15) * PS_LD + kk * 32 + kb * 16]);
        v8h ahi = *reinterpret_cast<const v8h*>(&Ps[(lane & 15) * PS_LD + kk * 32 + kb * 16 + 8]);
        v16h a = cat16(alo, ahi);
#pragma unroll
        for (int g = 0; g < 4; ++g) {
          const int e = wave * 64 + g * 16 + (lane & 15);
          v8h blo = *reinterpret_cast<const v8h*>(&Vst[e * VST_LD + kk * 32 + kb * 16]);
          v8h bhi = *reinterpret_cast<const v8h*>(&Vst[e * VST_LD + kk * 32 + kb * 16 + 8]);
          C[g] = wmma16(a, cat16(blo, bhi), C[g]);
        }
      }
    }
    __syncthreads();   // protect Vst/Ss before next block's staging
  }

  // ---- finalize: divide by row sums, write into concat[b, s, h*512 + e] (f16) ----
  float li[8];
#pragma unroll
  for (int vv = 0; vv < 8; ++vv) li[vv] = 1.0f / l_run[vv + 8 * (lane >> 4)];
#pragma unroll
  for (int g = 0; g < 4; ++g) {
    const int col = h * D_ + wave * 64 + g * 16 + (lane & 15);
#pragma unroll
    for (int vv = 0; vv < 8; ++vv) {
      const int row = qbase + vv + 8 * (lane >> 4);
      concat[((size_t)(b * S_ + row)) * HD_ + col] = (h16)(C[g][vv] * li[vv]);
    }
  }
}

// ---------------------------------------------------------------------------
// Kernel 3: output projection  out = concat(f16, 4096x4096) @ Wo(4096x512) + bo
// A tile is a pure f16 copy -> staged with async-to-LDS (ASYNCcnt).
// ---------------------------------------------------------------------------
__global__ __launch_bounds__(256) void out_gemm(
    const h16* __restrict__ A, const float* __restrict__ Wo,
    const float* __restrict__ bo, float* __restrict__ out)
{
  constexpr int BM = 128, BN = 128, BK = 32, PAD = 8;
  __shared__ alignas(16) h16 As[BM][BK + PAD];
  __shared__ alignas(16) h16 Bs[BN][BK + PAD];

  const int tid = threadIdx.x, wave = tid >> 5, lane = tid & 31;
  const int bm = blockIdx.x * BM, bn = blockIdx.y * BN;

  v8f zero = {};
  v8f C[8];
#pragma unroll
  for (int g = 0; g < 8; ++g) C[g] = zero;

  for (int kk = 0; kk < HD_; kk += BK) {
    if (kk + BK < HD_) {
      __builtin_prefetch(&A[(size_t)(bm + (tid & 127)) * HD_ + kk + BK], 0, 1);
      if (tid < BK) __builtin_prefetch(&Wo[(size_t)(kk + BK + tid) * D_ + bn], 0, 1);
    }
    // stage A via async copy (16B chunks, f16 -> f16)
    for (int idx = tid; idx < BM * (BK / 8); idx += 256) {
      const int r = idx >> 2, c = (idx & 3) * 8;
      async_ld_b128(&As[r][c], &A[(size_t)(bm + r) * HD_ + kk + c]);
    }
    // stage B transposed (f32 -> f16)
    for (int idx = tid; idx < BK * (BN / 4); idx += 256) {
      const int kr = idx >> 5, n = (idx & 31) * 4;
      float4 f = *reinterpret_cast<const float4*>(&Wo[(size_t)(kk + kr) * D_ + bn + n]);
      Bs[n + 0][kr] = (h16)f.x; Bs[n + 1][kr] = (h16)f.y;
      Bs[n + 2][kr] = (h16)f.z; Bs[n + 3][kr] = (h16)f.w;
    }
    wait_asynccnt0();
    __syncthreads();

    const int mr = wave * 16 + (lane & 15), kb = lane >> 4;
    v8h alo = *reinterpret_cast<const v8h*>(&As[mr][kb * 16]);
    v8h ahi = *reinterpret_cast<const v8h*>(&As[mr][kb * 16 + 8]);
    v16h a = cat16(alo, ahi);
#pragma unroll
    for (int g = 0; g < 8; ++g) {
      const int nr = g * 16 + (lane & 15), kg = (lane >> 4) * 16;
      v8h blo = *reinterpret_cast<const v8h*>(&Bs[nr][kg]);
      v8h bhi = *reinterpret_cast<const v8h*>(&Bs[nr][kg + 8]);
      C[g] = wmma16(a, cat16(blo, bhi), C[g]);
    }
    __syncthreads();
  }

  const int ml = 8 * (lane >> 4);
#pragma unroll
  for (int g = 0; g < 8; ++g) {
    const int col = bn + g * 16 + (lane & 15);
    const float bias = bo[col];
#pragma unroll
    for (int vv = 0; vv < 8; ++vv) {
      const int row = bm + wave * 16 + ml + vv;
      out[(size_t)row * D_ + col] = C[g][vv] + bias;
    }
  }
}

// ---------------------------------------------------------------------------
extern "C" void kernel_launch(void* const* d_in, const int* in_sizes, int n_in,
                              void* d_out, int out_size, void* d_ws, size_t ws_size,
                              hipStream_t stream) {
  (void)in_sizes; (void)n_in; (void)out_size; (void)ws_size;

  const float* q  = (const float*)d_in[0];
  const float* k  = (const float*)d_in[1];
  const float* v  = (const float*)d_in[2];
  const float* Wq = (const float*)d_in[3];
  const float* Wk = (const float*)d_in[4];
  const float* Wv = (const float*)d_in[5];
  const float* bq = (const float*)d_in[6];
  const float* bk = (const float*)d_in[7];
  const float* bv = (const float*)d_in[8];
  const float* Wo = (const float*)d_in[9];
  const float* bo = (const float*)d_in[10];

  // workspace: f16 Q,K,V ([H][B*S][D]) + f16 concat ([B*S][H*D]) = 128 MiB
  h16* Qh = (h16*)d_ws;
  h16* Kh = Qh + (size_t)H_ * M_ * D_;
  h16* Vh = Kh + (size_t)H_ * M_ * D_;
  h16* concat = Vh + (size_t)H_ * M_ * D_;

  proj_gemm<<<dim3(M_ / 128, D_ / 128, 3 * H_), 256, 0, stream>>>(
      q, k, v, Wq, Wk, Wv, bq, bk, bv, Qh, Kh, Vh);

  const size_t smem = 16640 + 139264 + 8448 + 4352 + 1024 + 192;  // 169,920 B (<320KB/WGP)
  attn_flash<<<dim3(S_ / 16, H_ * B_), 256, smem, stream>>>(Qh, Kh, Vh, concat);

  out_gemm<<<dim3(M_ / 128, D_ / 128), 256, 0, stream>>>(concat, Wo, bo, (float*)d_out);
}
